// MultiHeadGATLayer_86260123174621
// MI455X (gfx1250) — matrix operand order repaired and checked by reading, not compiled, not run on hardware
//
#include <hip/hip_runtime.h>
#include <hip/hip_bf16.h>
#include <math.h>

typedef __attribute__((ext_vector_type(16))) _Float16 v16h;
typedef __attribute__((ext_vector_type(8)))  float    v8f;

#define NHEAD 8
#define HDIM  32
#define IND   256   // IN_DIM == H*D

// ---------------------------------------------------------------------------
// float atomic max via int/uint atomics (valid with -inf initialization)
// ---------------------------------------------------------------------------
__device__ __forceinline__ void atomicMaxFloat(float* addr, float val) {
    if (val >= 0.0f) {
        atomicMax((int*)addr, __float_as_int(val));
    } else {
        atomicMin((unsigned int*)addr, __float_as_uint(val));
    }
}

// ---------------------------------------------------------------------------
// 0) Convert W [256(k),256(j)] f32 row-major -> WhT [j,k] f16 (B operand)
// ---------------------------------------------------------------------------
__global__ void gat_convW(const float* __restrict__ W, _Float16* __restrict__ WhT) {
    int idx = blockIdx.x * blockDim.x + threadIdx.x;   // idx = j*256 + k
    int j = idx >> 8;
    int k = idx & 255;
    WhT[idx] = (_Float16)W[k * IND + j];
}

// ---------------------------------------------------------------------------
// 1) Init: zero accumulator (d_out), m = -inf, denom = 0
// ---------------------------------------------------------------------------
__global__ void gat_init(float* __restrict__ out, float* __restrict__ mmax,
                         float* __restrict__ denom, int N) {
    int idx = blockIdx.x * blockDim.x + threadIdx.x;
    if (idx < N * IND) out[idx] = 0.0f;
    if (idx < N * NHEAD) {
        mmax[idx]  = -__builtin_inff();
        denom[idx] = 0.0f;
    }
}

// ---------------------------------------------------------------------------
// 2) GEMM hh = x @ W via v_wmma_f32_16x16x32_f16.
//    One wave (32 lanes) computes a 16x16 tile of hh. grid = (N/16, 16).
// ---------------------------------------------------------------------------
__global__ void gat_gemm_wmma(const float* __restrict__ x,
                              const _Float16* __restrict__ WhT,
                              float* __restrict__ hh) {
    const int rt   = blockIdx.x;        // row tile (M)
    const int ct   = blockIdx.y;        // col tile (N), 0..15
    const int lane = threadIdx.x;       // 0..31 (wave32)
    const int hi   = lane >> 4;         // lane group 0/1
    const int mn   = lane & 15;

    const float*    xrow = x   + (size_t)(rt * 16 + mn) * IND;   // A: row M=mn
    const _Float16* wcol = WhT + (size_t)(ct * 16 + mn) * IND;   // B: col N=mn

    const int akb = hi * 8;    // A K-base within each 16-half group
    const int bkb = hi * 16;   // unused; B uses akb (same K order as A)

    v8f c = {};
#pragma unroll
    for (int k0 = 0; k0 < IND; k0 += 32) {
        v16h a, b;
#pragma unroll
        for (int i = 0; i < 8; ++i) {
            a[i]     = (_Float16)xrow[k0 + akb + i];        // K = akb+i
            a[8 + i] = (_Float16)xrow[k0 + 16 + akb + i];   // K = 16+akb+i
        }
#pragma unroll
        for (int i = 0; i < 16; ++i) {
            b[i] = wcol[k0 + (i < 8 ? akb + i : 16 + akb + (i - 8))];   // K matches A
        }
        c = __builtin_amdgcn_wmma_f32_16x16x32_f16(
                /*neg_a=*/false, a, /*neg_b=*/false, b,
                /*c_mod=*/(short)0, c, /*reuse_a=*/false, /*reuse_b=*/false);
    }

    // C/D layout: VGPR r -> M = hi*8 + r, N = mn
    float* obase = hh + (size_t)(rt * 16) * IND + ct * 16 + mn;
#pragma unroll
    for (int r = 0; r < 8; ++r) {
        obase[(size_t)(hi * 8 + r) * IND] = c[r];
    }
}

// ---------------------------------------------------------------------------
// 3) Per-(node,head) score dots: s_src = hh.a_src, s_dst = hh.a_dst
// ---------------------------------------------------------------------------
__global__ void gat_head_scores(const float* __restrict__ hh,
                                const float* __restrict__ a_src,
                                const float* __restrict__ a_dst,
                                float* __restrict__ ssrc,
                                float* __restrict__ sdst, int N) {
    int idx = blockIdx.x * blockDim.x + threadIdx.x;   // n*8 + h
    if (idx >= N * NHEAD) return;
    int h = idx & (NHEAD - 1);
    const float* hp = hh + (size_t)(idx >> 3) * IND + h * HDIM;
    const float* as = a_src + h * HDIM;
    const float* ad = a_dst + h * HDIM;
    float s1 = 0.0f, s2 = 0.0f;
#pragma unroll
    for (int d = 0; d < HDIM; ++d) {
        float v = hp[d];
        s1 += v * as[d];
        s2 += v * ad[d];
    }
    ssrc[idx] = s1;
    sdst[idx] = s2;
}

// ---------------------------------------------------------------------------
// 4) Edge scores: leaky_relu(s_src[src]+s_dst[dst]); segment max into mmax
// ---------------------------------------------------------------------------
__global__ void gat_edge_score(const int* __restrict__ ei,
                               const float* __restrict__ ssrc,
                               const float* __restrict__ sdst,
                               float* __restrict__ scores,
                               float* __restrict__ mmax, int E) {
    int e = blockIdx.x * blockDim.x + threadIdx.x;
    if (e >= E) return;
    int s = ei[e];
    int d = ei[E + e];
#pragma unroll
    for (int h = 0; h < NHEAD; ++h) {
        float sc = ssrc[s * NHEAD + h] + sdst[d * NHEAD + h];
        sc = (sc > 0.0f) ? sc : 0.2f * sc;   // leaky_relu(0.2)
        scores[(size_t)e * NHEAD + h] = sc;
        atomicMaxFloat(&mmax[d * NHEAD + h], sc);
    }
}

// ---------------------------------------------------------------------------
// 5) exp(score - max) ; segment sum into denom (scores overwritten with e)
// ---------------------------------------------------------------------------
__global__ void gat_edge_exp(const int* __restrict__ ei,
                             float* __restrict__ scores,
                             const float* __restrict__ mmax,
                             float* __restrict__ denom, int E) {
    int e = blockIdx.x * blockDim.x + threadIdx.x;
    if (e >= E) return;
    int d = ei[E + e];
#pragma unroll
    for (int h = 0; h < NHEAD; ++h) {
        float ex = __expf(scores[(size_t)e * NHEAD + h] - mmax[d * NHEAD + h]);
        scores[(size_t)e * NHEAD + h] = ex;
        atomicAdd(&denom[d * NHEAD + h], ex);
    }
}

// ---------------------------------------------------------------------------
// 6) Weighted scatter-add: out[dst] += alpha * hh[src]. One 256-thread
//    block per edge; thread t handles (h=t>>5, d=t&31). hh sits in L2.
// ---------------------------------------------------------------------------
__global__ void gat_aggregate(const int* __restrict__ ei,
                              const float* __restrict__ hh,
                              const float* __restrict__ scores,
                              const float* __restrict__ denom,
                              float* __restrict__ out, int E) {
    int e = blockIdx.x;
    int t = threadIdx.x;            // 0..255
    int h = t >> 5;
    int s = ei[e];
    int d = ei[E + e];
    float alpha = scores[(size_t)e * NHEAD + h] / denom[d * NHEAD + h];
    float val   = hh[(size_t)s * IND + t] * alpha;
    atomicAdd(&out[(size_t)d * IND + t], val);
}

// ---------------------------------------------------------------------------
// 7) Finalize: residual + LayerNorm + affine + ELU. One wave32 per node,
//    8 elements/lane, shuffle reduction (wave32-correct).
// ---------------------------------------------------------------------------
__global__ void gat_finalize(const float* __restrict__ x,
                             const float* __restrict__ gamma,
                             const float* __restrict__ beta,
                             float* __restrict__ out, int N) {
    int n = blockIdx.x * (blockDim.x >> 5) + (threadIdx.x >> 5);
    if (n >= N) return;
    int lane = threadIdx.x & 31;

    float v[8];
    float sum = 0.0f, sq = 0.0f;
    size_t base = (size_t)n * IND + lane * 8;
#pragma unroll
    for (int i = 0; i < 8; ++i) {
        float t = out[base + i] + x[base + i];   // residual
        v[i] = t;
        sum += t;
        sq  += t * t;
    }
#pragma unroll
    for (int o = 16; o > 0; o >>= 1) {
        sum += __shfl_xor(sum, o, 32);
        sq  += __shfl_xor(sq,  o, 32);
    }
    float mu   = sum * (1.0f / IND);
    float var  = sq * (1.0f / IND) - mu * mu;
    float rstd = rsqrtf(var + 1e-5f);
#pragma unroll
    for (int i = 0; i < 8; ++i) {
        int j = lane * 8 + i;
        float y = (v[i] - mu) * rstd * gamma[j] + beta[j];
        out[base + i] = (y > 0.0f) ? y : (__expf(y) - 1.0f);
    }
}

// ---------------------------------------------------------------------------
extern "C" void kernel_launch(void* const* d_in, const int* in_sizes, int n_in,
                              void* d_out, int out_size, void* d_ws, size_t ws_size,
                              hipStream_t stream) {
    const float* x     = (const float*)d_in[0];
    const int*   ei    = (const int*)  d_in[1];
    const float* W     = (const float*)d_in[2];
    const float* a_src = (const float*)d_in[3];
    const float* a_dst = (const float*)d_in[4];
    const float* gamma = (const float*)d_in[5];
    const float* beta  = (const float*)d_in[6];
    float* out = (float*)d_out;

    const int N = in_sizes[0] / IND;    // 50000
    const int E = in_sizes[1] / 2;      // 400000

    // workspace layout
    char* ws = (char*)d_ws;
    _Float16* WhT = (_Float16*)ws;                       // 256*256*2 B
    float* hh     = (float*)(ws + (size_t)IND * IND * 2);
    float* ssrc   = hh    + (size_t)N * IND;
    float* sdst   = ssrc  + (size_t)N * NHEAD;
    float* mmax   = sdst  + (size_t)N * NHEAD;
    float* denom  = mmax  + (size_t)N * NHEAD;
    float* scores = denom + (size_t)N * NHEAD;           // E*8 floats

    // 0) transpose+convert W to f16
    gat_convW<<<(IND * IND) / 256, 256, 0, stream>>>(W, WhT);
    // 1) init accumulators
    gat_init<<<(N * IND + 255) / 256, 256, 0, stream>>>(out, mmax, denom, N);
    // 2) WMMA GEMM (N divisible by 16: 50000/16 = 3125)
    dim3 ggrid(N / 16, IND / 16);
    gat_gemm_wmma<<<ggrid, 32, 0, stream>>>(x, WhT, hh);
    // 3) per-(node,head) score dots
    gat_head_scores<<<(N * NHEAD + 255) / 256, 256, 0, stream>>>(hh, a_src, a_dst,
                                                                 ssrc, sdst, N);
    // 4) edge scores + segment max
    gat_edge_score<<<(E + 255) / 256, 256, 0, stream>>>(ei, ssrc, sdst, scores,
                                                        mmax, E);
    // 5) exp + segment sum
    gat_edge_exp<<<(E + 255) / 256, 256, 0, stream>>>(ei, scores, mmax, denom, E);
    // 6) weighted scatter-add aggregation
    gat_aggregate<<<E, IND, 0, stream>>>(ei, hh, scores, denom, out, E);
    // 7) residual + LayerNorm + ELU (8 wave32 per block)
    gat_finalize<<<(N + 7) / 8, 256, 0, stream>>>(x, gamma, beta, out, N);
}
